// snnLinear_80058190397921
// MI455X (gfx1250) — 8-bit matrix operand order repaired and checked by reading, compiled, not run on hardware
//
#include <hip/hip_runtime.h>
#include <stdint.h>

typedef int v8i __attribute__((ext_vector_type(8)));

#define IN_F   4096
#define OUT_F  4096
#define M_TOT  8192

__device__ __forceinline__ void fwht_lds(float* lds) {
    for (int len = 1; len < IN_F; len <<= 1) {
        for (int p = threadIdx.x; p < IN_F / 2; p += 256) {
            int i = ((p & ~(len - 1)) << 1) | (p & (len - 1));
            float a = lds[i];
            float b = lds[i + len];
            lds[i]       = a + b;
            lds[i + len] = a - b;
        }
        __syncthreads();
    }
}

__global__ __launch_bounds__(256)
void fwht_minmax_kernel(const float* __restrict__ x,
                        float* __restrict__ blkmin,
                        float* __restrict__ blkmax) {
    __shared__ float lds[IN_F];
    __shared__ float smin[256];
    __shared__ float smax[256];

    const int row = blockIdx.x;
    const float* xr = x + (size_t)row * IN_F;
    for (int i = threadIdx.x; i < IN_F; i += 256) lds[i] = xr[i];
    __syncthreads();

    fwht_lds(lds);

    float mn =  3.0e38f, mx = -3.0e38f;
    for (int i = threadIdx.x; i < IN_F; i += 256) {
        float v = lds[i] * (1.0f / 64.0f);
        mn = fminf(mn, v);
        mx = fmaxf(mx, v);
    }
    smin[threadIdx.x] = mn;
    smax[threadIdx.x] = mx;
    __syncthreads();
    for (int s = 128; s > 0; s >>= 1) {
        if ((int)threadIdx.x < s) {
            smin[threadIdx.x] = fminf(smin[threadIdx.x], smin[threadIdx.x + s]);
            smax[threadIdx.x] = fmaxf(smax[threadIdx.x], smax[threadIdx.x + s]);
        }
        __syncthreads();
    }
    if (threadIdx.x == 0) {
        blkmin[row] = smin[0];
        blkmax[row] = smax[0];
    }
}

__global__ __launch_bounds__(256)
void minmax_reduce_kernel(const float* __restrict__ blkmin,
                          const float* __restrict__ blkmax,
                          float* __restrict__ qparams, int nrows) {
    __shared__ float smin[256];
    __shared__ float smax[256];
    float mn =  3.0e38f, mx = -3.0e38f;
    for (int i = threadIdx.x; i < nrows; i += 256) {
        mn = fminf(mn, blkmin[i]);
        mx = fmaxf(mx, blkmax[i]);
    }
    smin[threadIdx.x] = mn;
    smax[threadIdx.x] = mx;
    __syncthreads();
    for (int s = 128; s > 0; s >>= 1) {
        if ((int)threadIdx.x < s) {
            smin[threadIdx.x] = fminf(smin[threadIdx.x], smin[threadIdx.x + s]);
            smax[threadIdx.x] = fmaxf(smax[threadIdx.x], smax[threadIdx.x + s]);
        }
        __syncthreads();
    }
    if (threadIdx.x == 0) {
        float xmin = smin[0], xmax = smax[0];
        float scale = fmaxf((xmax - xmin) * (1.0f / 15.0f), 1e-8f);
        float zp = rintf(-xmin / scale);
        qparams[0] = scale;
        qparams[1] = zp;
    }
}

__global__ __launch_bounds__(256)
void fwht_quant_kernel(const float* __restrict__ x,
                       const float* __restrict__ qparams,
                       uint8_t* __restrict__ qbuf) {
    __shared__ float lds[IN_F];
    const int row = blockIdx.x;
    const float* xr = x + (size_t)row * IN_F;
    for (int i = threadIdx.x; i < IN_F; i += 256) lds[i] = xr[i];
    __syncthreads();

    fwht_lds(lds);

    const float inv_scale = 1.0f / qparams[0];
    const float zp = qparams[1];
    uint8_t* qr = qbuf + (size_t)row * IN_F;
    for (int k = threadIdx.x; k < IN_F; k += 256) {
        float v = lds[k] * (1.0f / 64.0f);
        float qf = rintf(v * inv_scale) + zp;
        qf = fminf(fmaxf(qf, 0.0f), 15.0f);
        const int g    = (k >> 5) & 1;
        const int half = (k >> 4) & 1;
        const int off8 = k & 15;
        const int pos  = (k & ~63) + half * 32 + g * 16 + off8;
        qr[pos] = (uint8_t)qf;
    }
}

__global__ __launch_bounds__(256)
void wquant_kernel(const float* __restrict__ w,
                   int8_t* __restrict__ qw,
                   float* __restrict__ wscale,
                   int* __restrict__ wrowsum) {
    __shared__ float sred[256];
    __shared__ int   isum[256];
    const int o = blockIdx.x;
    const float* wr = w + (size_t)o * IN_F;

    float amax = 0.0f;
    for (int i = threadIdx.x; i < IN_F; i += 256) amax = fmaxf(amax, fabsf(wr[i]));
    sred[threadIdx.x] = amax;
    __syncthreads();
    for (int s = 128; s > 0; s >>= 1) {
        if ((int)threadIdx.x < s)
            sred[threadIdx.x] = fmaxf(sred[threadIdx.x], sred[threadIdx.x + s]);
        __syncthreads();
    }
    const float scale = fmaxf(sred[0] * (1.0f / 7.0f), 1e-8f);
    const float inv_scale = 1.0f / scale;
    __syncthreads();

    int rsum = 0;
    int8_t* qr = qw + (size_t)o * IN_F;
    for (int i = threadIdx.x; i < IN_F; i += 256) {
        float qf = fminf(fmaxf(rintf(wr[i] * inv_scale), -8.0f), 7.0f);
        qr[i] = (int8_t)qf;
        rsum += (int)qf;
    }
    isum[threadIdx.x] = rsum;
    __syncthreads();
    for (int s = 128; s > 0; s >>= 1) {
        if ((int)threadIdx.x < s) isum[threadIdx.x] += isum[threadIdx.x + s];
        __syncthreads();
    }
    if (threadIdx.x == 0) {
        wscale[o] = scale;
        wrowsum[o] = isum[0];
    }
}

__device__ __forceinline__ v8i load_a_frag(const uint8_t* arow_k, int half) {
    const int4* p = (const int4*)(arow_k + half * 32);
    int4 lo = p[0];
    int4 hi = p[1];
    v8i a;
    a[0] = lo.x; a[1] = lo.y; a[2] = lo.z; a[3] = lo.w;
    a[4] = hi.x; a[5] = hi.y; a[6] = hi.z; a[7] = hi.w;
    return a;
}

__device__ __forceinline__ v8i load_b_frag(const int8_t* brow, int kh) {
    const int4* p = (const int4*)brow;
    int4 lo = p[kh];
    int4 hi = p[kh + 2];
    v8i b;
    b[0] = lo.x; b[1] = lo.y; b[2] = lo.z; b[3] = lo.w;
    b[4] = hi.x; b[5] = hi.y; b[6] = hi.z; b[7] = hi.w;
    return b;
}

__global__ __launch_bounds__(256)
void wmma_gemm_kernel(const uint8_t* __restrict__ q,
                      const int8_t*  __restrict__ qw,
                      const float*   __restrict__ wscale,
                      const int*     __restrict__ wrowsum,
                      const float*   __restrict__ bias,
                      const float*   __restrict__ qparams,
                      float*         __restrict__ out) {
    const int lane = threadIdx.x & 31;
    const int wave = threadIdx.x >> 5;
    const int wm = wave & 3;
    const int wn = wave >> 2;
    const int m_base = blockIdx.x * 128 + wm * 32;
    const int n_base = blockIdx.y * 128 + wn * 64;
    const int mrow = lane & 15;
    const int half = lane >> 4;

    const uint8_t* arow0 = q + (size_t)(m_base + mrow) * IN_F;
    const uint8_t* arow1 = arow0 + (size_t)16 * IN_F;
    const int8_t*  brow0 = qw + (size_t)(n_base + 0  + mrow) * IN_F;
    const int8_t*  brow1 = qw + (size_t)(n_base + 16 + mrow) * IN_F;
    const int8_t*  brow2 = qw + (size_t)(n_base + 32 + mrow) * IN_F;
    const int8_t*  brow3 = qw + (size_t)(n_base + 48 + mrow) * IN_F;

    v8i acc[8];
#pragma unroll
    for (int i = 0; i < 8; ++i) acc[i] = (v8i){0,0,0,0,0,0,0,0};

    for (int k0 = 0; k0 < IN_F; k0 += 64) {
        v8i a0 = load_a_frag(arow0 + k0, half);
        v8i a1 = load_a_frag(arow1 + k0, half);
        v8i b0 = load_b_frag(brow0 + k0, half);
        v8i b1 = load_b_frag(brow1 + k0, half);
        v8i b2 = load_b_frag(brow2 + k0, half);
        v8i b3 = load_b_frag(brow3 + k0, half);
        acc[0] = __builtin_amdgcn_wmma_i32_16x16x64_iu8(false, a0, true, b0, acc[0], false, false);
        acc[1] = __builtin_amdgcn_wmma_i32_16x16x64_iu8(false, a0, true, b1, acc[1], false, false);
        acc[2] = __builtin_amdgcn_wmma_i32_16x16x64_iu8(false, a0, true, b2, acc[2], false, false);
        acc[3] = __builtin_amdgcn_wmma_i32_16x16x64_iu8(false, a0, true, b3, acc[3], false, false);
        acc[4] = __builtin_amdgcn_wmma_i32_16x16x64_iu8(false, a1, true, b0, acc[4], false, false);
        acc[5] = __builtin_amdgcn_wmma_i32_16x16x64_iu8(false, a1, true, b1, acc[5], false, false);
        acc[6] = __builtin_amdgcn_wmma_i32_16x16x64_iu8(false, a1, true, b2, acc[6], false, false);
        acc[7] = __builtin_amdgcn_wmma_i32_16x16x64_iu8(false, a1, true, b3, acc[7], false, false);
    }

    const float sx = qparams[0];
    const float zp = qparams[1];
#pragma unroll
    for (int t = 0; t < 4; ++t) {
        const int n = n_base + t * 16 + mrow;
        const float sw   = wscale[n];
        const float corr = zp * (float)wrowsum[n];
        const float bs   = bias[n];
        const float ds   = sx * sw;
#pragma unroll
        for (int ms = 0; ms < 2; ++ms) {
            const v8i a = acc[ms * 4 + t];
#pragma unroll
            for (int r = 0; r < 8; ++r) {
                const int m = m_base + ms * 16 + half * 8 + r;
                out[(size_t)m * OUT_F + n] = ds * ((float)a[r] - corr) + bs;
            }
        }
    }
}

extern "C" void kernel_launch(void* const* d_in, const int* in_sizes, int n_in,
                              void* d_out, int out_size, void* d_ws, size_t ws_size,
                              hipStream_t stream) {
    const float* x      = (const float*)d_in[0];
    const float* weight = (const float*)d_in[1];
    const float* bias   = (const float*)d_in[2];
    float* out = (float*)d_out;

    uint8_t* base = (uint8_t*)d_ws;
    float* blkmin  = (float*)(base);
    float* blkmax  = (float*)(base + 8192 * 4);
    float* qparams = (float*)(base + 65536);
    float* wscale  = (float*)(base + 65536 + 1024);
    int*   wrowsum = (int*)  (base + 65536 + 1024 + OUT_F * 4);
    int8_t*  qw    = (int8_t*) (base + (size_t)(1 << 20));
    uint8_t* qbuf  = (uint8_t*)(base + (size_t)(1 << 20) + (16u << 20));

    fwht_minmax_kernel<<<M_TOT, 256, 0, stream>>>(x, blkmin, blkmax);
    minmax_reduce_kernel<<<1, 256, 0, stream>>>(blkmin, blkmax, qparams, M_TOT);
    fwht_quant_kernel<<<M_TOT, 256, 0, stream>>>(x, qparams, qbuf);
    wquant_kernel<<<OUT_F, 256, 0, stream>>>(weight, qw, wscale, wrowsum);

    dim3 grid(M_TOT / 128, OUT_F / 128);
    wmma_gemm_kernel<<<grid, 256, 0, stream>>>(qbuf, qw, wscale, wrowsum,
                                               bias, qparams, out);
}
